// GPTModel_5162550690268
// MI455X (gfx1250) — matrix operand order repaired and checked by reading, not compiled, not run on hardware
//
#include <hip/hip_runtime.h>
#include <hip/hip_bf16.h>

typedef __attribute__((ext_vector_type(16))) _Float16 v16h;
typedef __attribute__((ext_vector_type(8)))  _Float16 v8h;
typedef __attribute__((ext_vector_type(8)))  float    v8f;

#define BB   2
#define TT   1024
#define BT   (BB*TT)        // 2048
#define DD   768
#define HH   12
#define HSZ  64
#define LLAY 4
#define DFF  (4*DD)         // 3072
#define VV   32000
#define EPSLN 1e-5f
#define ATT_SCALE 0.03608439182435161f   // 768^-0.5 (reference scales by emb_dim)

// gfx1250 wave32 WMMA fragment maps (CDNA5 ISA):
//  A (16x32 f16): lane m=lane&15; elem j -> K = (j&7) + 16*(j>>3) + 8*(lane>>4)
//  B (32x16 f16): lane n=lane&15; elem j -> K = (j&7) + 16*(j>>3) + 8*(lane>>4)
//  C/D (16x16 f32): elem r -> M = r + 8*(lane>>4), N = lane&15

__device__ __forceinline__ v16h ldpair(const _Float16* lo, const _Float16* hi) {
  v8h a = *(const v8h*)lo;          // 16B vector load
  v8h b = *(const v8h*)hi;          // 16B vector load
  return __builtin_shufflevector(a, b, 0,1,2,3,4,5,6,7,8,9,10,11,12,13,14,15);
}

__device__ __forceinline__ v8f wmma32(v16h a, v16h b, v8f c) {
  return __builtin_amdgcn_wmma_f32_16x16x32_f16(false, a, false, b,
                                                (short)0, c, false, false);
}

// =============================== embedding ==================================
__global__ __launch_bounds__(256) void embed_kernel(
    const int* __restrict__ tokens, const float* __restrict__ tok_emb,
    const float* __restrict__ pos_emb, float* __restrict__ x)
{
  const int i   = blockIdx.x;            // [0, BT)
  const int t   = i & (TT - 1);
  const int tok = tokens[i];
  const float* pe = tok_emb + (size_t)tok * DD;
  const float* pp = pos_emb + (size_t)t  * DD;
  float* px = x + (size_t)i * DD;
  for (int c = threadIdx.x; c < DD; c += 256) px[c] = pe[c] + pp[c];
}

// =============================== layernorm ==================================
__global__ __launch_bounds__(256) void ln_kernel(
    const float* __restrict__ x, const float* __restrict__ g,
    const float* __restrict__ bta, _Float16* __restrict__ out)
{
  __shared__ float red[256];
  __shared__ float stat[2];
  const int row = blockIdx.x;
  const int tid = threadIdx.x;
  const float* px = x + (size_t)row * DD;

  float s = 0.f;
  for (int c = tid; c < DD; c += 256) s += px[c];
  red[tid] = s; __syncthreads();
  for (int o = 128; o; o >>= 1) { if (tid < o) red[tid] += red[tid + o]; __syncthreads(); }
  if (tid == 0) stat[0] = red[0] * (1.0f / DD);
  __syncthreads();
  const float mu = stat[0];

  float s2 = 0.f;
  for (int c = tid; c < DD; c += 256) { float d = px[c] - mu; s2 += d * d; }
  red[tid] = s2; __syncthreads();
  for (int o = 128; o; o >>= 1) { if (tid < o) red[tid] += red[tid + o]; __syncthreads(); }
  if (tid == 0) stat[1] = rsqrtf(red[0] * (1.0f / DD) + EPSLN);
  __syncthreads();
  const float rstd = stat[1];

  _Float16* po = out + (size_t)row * DD;
  for (int c = tid; c < DD; c += 256)
    po[c] = (_Float16)((px[c] - mu) * rstd * g[c] + bta[c]);
}

// ================================ GEMM ======================================
// C[M,N] = epilogue( A16[M,K] x Bf32[K,N] ).  128x128 block tile, 4 waves,
// K-tile 64 deep (2 x 32 sub-steps per barrier pair), 32 WMMAs per K-tile per
// wave.  B fragments are software-rotated (load nt+1 before consuming nt) so
// every DS load has a full WMMA pair + address math of slack.
// B column mapping addr(k,col) = k*ldb + (col>>colShift)*colStride + (col&mask)
// handles row-major [K,N] (shift=30) and the [H,D,HS] QKV layout (shift=6).
// B is stored transposed in LDS ([N][K], pitch 72) -> B frags are 2x b128.
// vTrans=1 writes f16 output as [B,H,HS,T] (pre-transposed V for attention).
#define GTM 128
#define GTN 128
#define GTK 64
#define LD_S 72    // halves; 144B pitch (16B aligned)

__global__ __launch_bounds__(128) void gemm_wmma_kernel(
    const _Float16* __restrict__ A, const float* __restrict__ Bsrc,
    int Kdim, int Ndim, int colShift, int colStride, int ldb,
    const float* __restrict__ bias, const float* __restrict__ residual,
    float* __restrict__ outF32, _Float16* __restrict__ outF16,
    int applyGelu, int vTrans)
{
  __shared__ _Float16 As[GTM * LD_S];   // [row][k]
  __shared__ _Float16 Bs[GTN * LD_S];   // [col][k]  (transposed)
  const int tid  = threadIdx.x;
  const int lane = tid & 31;
  const int wave = tid >> 5;
  const int col0 = blockIdx.x * GTN;
  const int row0 = blockIdx.y * GTM;
  const int colMask = (1 << colShift) - 1;
  const int m  = lane & 15;
  const int ho = (lane >> 4) << 3;   // A-fragment half offset
  const int ko = (lane >> 4) << 4;   // unused; B fragments also use ho

  v8f zero = {};
  v8f acc[2][8];
  #pragma unroll
  for (int i = 0; i < 2; ++i)
    #pragma unroll
    for (int nt = 0; nt < 8; ++nt) acc[i][nt] = zero;

  for (int k0 = 0; k0 < Kdim; k0 += GTK) {
    // A tile 128x64 halves: 1024 x 16B vector copies (8 per thread)
    for (int i = tid; i < (GTM * GTK) / 8; i += 128) {
      int r  = i >> 3;
      int c8 = (i & 7) << 3;
      *(v8h*)&As[r * LD_S + c8] =
          *(const v8h*)&A[(size_t)(row0 + r) * Kdim + k0 + c8];
    }
    // B tile 64x128 f32: float4 loads, cvt f16, transposed LDS store;
    // prefetch next K-tile (global_prefetch_b8) while WMMAs run.
    for (int i = tid; i < (GTK * GTN) / 4; i += 128) {
      int kk = i >> 5;
      int c4 = (i & 31) << 2;
      int col = col0 + c4;
      size_t baddr = (size_t)(k0 + kk) * ldb +
                     (size_t)(col >> colShift) * colStride + (col & colMask);
      float4 f = *(const float4*)&Bsrc[baddr];
      if (k0 + GTK < Kdim)
        __builtin_prefetch(&Bsrc[baddr + (size_t)GTK * ldb], 0, 1);
      Bs[(c4 + 0) * LD_S + kk] = (_Float16)f.x;
      Bs[(c4 + 1) * LD_S + kk] = (_Float16)f.y;
      Bs[(c4 + 2) * LD_S + kk] = (_Float16)f.z;
      Bs[(c4 + 3) * LD_S + kk] = (_Float16)f.w;
    }
    __syncthreads();

    #pragma unroll
    for (int kt = 0; kt < 2; ++kt) {
      const int kb = kt * 32;
      v16h af0, af1;
      {
        const _Float16* pA = &As[(wave * 32 + m) * LD_S + kb];
        af0 = ldpair(pA + ho, pA + 16 + ho);
        pA += 16 * LD_S;
        af1 = ldpair(pA + ho, pA + 16 + ho);
      }
      // rotated B pipeline: next fragment's loads issue before current's WMMAs
      const _Float16* pB0 = &Bs[m * LD_S + kb + ho];
      v16h bcur = ldpair(pB0, pB0 + 16);
      #pragma unroll
      for (int nt = 0; nt < 8; ++nt) {
        v16h bnext = bcur;
        if (nt < 7) {
          const _Float16* pB = &Bs[((nt + 1) * 16 + m) * LD_S + kb + ho];
          bnext = ldpair(pB, pB + 16);
        }
        acc[0][nt] = wmma32(af0, bcur, acc[0][nt]);
        acc[1][nt] = wmma32(af1, bcur, acc[1][nt]);
        bcur = bnext;
      }
    }
    __syncthreads();
  }

  // fused epilogue: bias / exact GELU / residual, f32 or f16 (opt. V-transpose)
  #pragma unroll
  for (int i = 0; i < 2; ++i) {
    #pragma unroll
    for (int nt = 0; nt < 8; ++nt) {
      #pragma unroll
      for (int r = 0; r < 8; ++r) {
        int grow = row0 + wave * 32 + i * 16 + r + ho;
        int gcol = col0 + nt * 16 + m;
        float vv = acc[i][nt][r];
        if (bias) vv += bias[gcol];
        if (applyGelu) vv = 0.5f * vv * (1.0f + erff(vv * 0.70710678118654752f));
        if (vTrans) {
          // out[b][h][e][t] ; grow = b*TT + t, gcol = h*64 + e
          size_t oi = ((((size_t)(grow / TT)) * HH + (gcol >> 6)) * HSZ
                       + (gcol & 63)) * TT + (grow & (TT - 1));
          outF16[oi] = (_Float16)vv;
        } else {
          size_t oi = (size_t)grow * Ndim + gcol;
          if (residual) vv += residual[oi];
          if (outF32) outF32[oi] = vv;
          if (outF16) outF16[oi] = (_Float16)vv;
        }
      }
    }
  }
}

// ============================== attention ===================================
// 4 independent waves per block; one wave per (b, h, 16-query tile).
// Computes S^T (A = K rows, B = Q^T) so the C layout puts each lane on one
// query column t = lane&15 with s-rows (ct*16 + r + 8*half) -- exactly the
// A-fragment index map needed for P in the P.V WMMA.  Softmax state is a
// per-lane scalar; cross-lane traffic is one shfl_xor(16) for max and sum
// plus 8 alpha broadcasts for the O-row rescale.  No LDS, no barriers.
// All K *and* V fragment loads for a chunk are issued up front: V's global
// latency hides behind the QK^T WMMAs and the softmax VALU work.
__global__ __launch_bounds__(128) void attn_kernel(
    const _Float16* __restrict__ q, const _Float16* __restrict__ k,
    const _Float16* __restrict__ vT, float* __restrict__ x)
{
  const int lane = threadIdx.x & 31;
  const int wave = threadIdx.x >> 5;
  const int t0   = (blockIdx.x * 4 + wave) * 16;
  const int h    = blockIdx.y;
  const int b    = blockIdx.z;
  const size_t rowBase = (size_t)b * TT;
  const int colBase = h * HSZ;
  const int m  = lane & 15;
  const int ho = (lane >> 4) << 3;
  const int ko = (lane >> 4) << 4;
  const int tcol = t0 + m;               // this lane's query column

  // Q^T B-fragments (col t = m, K = head dim), loaded once
  v16h bq0, bq1;
  {
    const _Float16* pq = &q[(rowBase + tcol) * DD + colBase];
    bq0 = ldpair(pq + ho,      pq + 16 + ho);
    bq1 = ldpair(pq + 32 + ho, pq + 48 + ho);
  }

  float mrun = -INFINITY, lrun = 0.f;    // per-lane softmax state (column tcol)
  v8f zero = {};
  v8f acc[4];
  #pragma unroll
  for (int i = 0; i < 4; ++i) acc[i] = zero;

  for (int s0 = 0; s0 <= t0 + 15; s0 += 32) {   // causal: s0 <= last query row
    // ---- issue ALL global fragment loads for this chunk up front ----
    v16h ak[2][2];                       // K rows (A-fragments), 2 s-subtiles
    #pragma unroll
    for (int ct = 0; ct < 2; ++ct) {
      const _Float16* pk = &k[(rowBase + s0 + ct * 16 + m) * DD + colBase];
      ak[ct][0] = ldpair(pk + ho,      pk + 16 + ho);
      ak[ct][1] = ldpair(pk + 32 + ho, pk + 48 + ho);
    }
    v16h bv[4];                          // V columns (B-fragments), used late
    #pragma unroll
    for (int et = 0; et < 4; ++et) {
      const _Float16* pv =
          &vT[(((size_t)b * HH + h) * HSZ + et * 16 + m) * TT + s0 + ho];
      bv[et] = ldpair(pv, pv + 16);
    }

    // ---- S^T chunk: rows s0..s0+31, col tcol ----
    v8f sc[2];
    #pragma unroll
    for (int ct = 0; ct < 2; ++ct) {
      v8f c = zero;
      c = wmma32(ak[ct][0], bq0, c);
      c = wmma32(ak[ct][1], bq1, c);
      #pragma unroll
      for (int r = 0; r < 8; ++r) {
        int sabs = s0 + ct * 16 + r + ho;
        float vv = c[r] * ATT_SCALE;
        sc[ct][r] = (sabs > tcol) ? -INFINITY : vv;
      }
    }
    // column max: in-lane over 16, then combine with partner lane (xor 16)
    float mloc = -INFINITY;
    #pragma unroll
    for (int ct = 0; ct < 2; ++ct)
      #pragma unroll
      for (int r = 0; r < 8; ++r) mloc = fmaxf(mloc, sc[ct][r]);
    mloc = fmaxf(mloc, __shfl_xor(mloc, 16, 32));

    float mn    = fmaxf(mrun, mloc);
    float alpha = expf(mrun - mn);       // 0 on first chunk
    mrun = mn;

    float psum = 0.f;
    #pragma unroll
    for (int ct = 0; ct < 2; ++ct)
      #pragma unroll
      for (int r = 0; r < 8; ++r) {
        sc[ct][r] = expf(sc[ct][r] - mn);   // masked -> 0
        psum += sc[ct][r];
      }
    psum += __shfl_xor(psum, 16, 32);
    lrun = lrun * alpha + psum;

    // rescale O accumulators: alpha per O-row (t = r + ho) via broadcast
    float arow[8];
    #pragma unroll
    for (int r = 0; r < 8; ++r) arow[r] = __shfl(alpha, r + ho, 32);
    #pragma unroll
    for (int et = 0; et < 4; ++et)
      #pragma unroll
      for (int r = 0; r < 8; ++r) acc[et][r] *= arow[r];

    // P A-fragment: pure in-lane relayout (sc[j>>3][j&7])
    v16h pa;
    #pragma unroll
    for (int j = 0; j < 16; ++j) pa[j] = (_Float16)sc[j >> 3][j & 7];

    #pragma unroll
    for (int et = 0; et < 4; ++et)
      acc[et] = wmma32(pa, bv[et], acc[et]);
  }

  // normalize (l per O-row via broadcast) and residual-add into x
  float lrow[8];
  #pragma unroll
  for (int r = 0; r < 8; ++r) lrow[r] = __shfl(lrun, r + ho, 32);
  #pragma unroll
  for (int et = 0; et < 4; ++et) {
    #pragma unroll
    for (int r = 0; r < 8; ++r) {
      size_t oi = (rowBase + t0 + r + ho) * DD + colBase + et * 16 + m;
      x[oi] += acc[et][r] / lrow[r];
    }
  }
}

// ================================ driver ====================================
extern "C" void kernel_launch(void* const* d_in, const int* in_sizes, int n_in,
                              void* d_out, int out_size, void* d_ws, size_t ws_size,
                              hipStream_t stream) {
  (void)in_sizes; (void)n_in; (void)out_size; (void)ws_size;
  const int*   tokens    = (const int*)  d_in[0];
  const float* token_emb = (const float*)d_in[1];
  const float* pos_emb   = (const float*)d_in[2];
  const float* Wq    = (const float*)d_in[3];
  const float* Wk    = (const float*)d_in[4];
  const float* Wv    = (const float*)d_in[5];
  const float* ln1_g = (const float*)d_in[6];
  const float* ln1_b = (const float*)d_in[7];
  const float* W1    = (const float*)d_in[8];
  const float* b1    = (const float*)d_in[9];
  const float* W2    = (const float*)d_in[10];
  const float* b2    = (const float*)d_in[11];
  const float* ln2_g = (const float*)d_in[12];
  const float* ln2_b = (const float*)d_in[13];
  const float* lnf_g = (const float*)d_in[14];
  const float* lnf_b = (const float*)d_in[15];
  const float* Wlm   = (const float*)d_in[16];
  const float* blm   = (const float*)d_in[17];
  float* out = (float*)d_out;

  // workspace layout (256B-aligned offsets), ~31.5 MB total
  char* ws = (char*)d_ws;
  float*    x    = (float*)ws;                      // BT*D   f32
  _Float16* h16  = (_Float16*)(ws + 6291456);       // BT*D   f16
  _Float16* q16  = (_Float16*)(ws + 9437184);       // BT*D   f16  [BT,768]
  _Float16* k16  = (_Float16*)(ws + 12582912);      // BT*D   f16  [BT,768]
  _Float16* vT16 = (_Float16*)(ws + 15728640);      // BT*D   f16  [B,H,HS,T]
  _Float16* g16  = (_Float16*)(ws + 18874368);      // BT*DFF f16

  embed_kernel<<<BT, 256, 0, stream>>>(tokens, token_emb, pos_emb, x);

  const size_t wqkvStride = (size_t)HH * DD * HSZ;  // per-layer [H,D,HS]
  for (int l = 0; l < LLAY; ++l) {
    ln_kernel<<<BT, 256, 0, stream>>>(x, ln1_g + l * DD, ln1_b + l * DD, h16);

    dim3 gQKV(DD / GTN, BT / GTM);
    gemm_wmma_kernel<<<gQKV, 128, 0, stream>>>(h16, Wq + l * wqkvStride,
        DD, DD, /*shift*/6, /*stride*/DD * HSZ, /*ldb*/HSZ,
        nullptr, nullptr, nullptr, q16, 0, 0);
    gemm_wmma_kernel<<<gQKV, 128, 0, stream>>>(h16, Wk + l * wqkvStride,
        DD, DD, 6, DD * HSZ, HSZ, nullptr, nullptr, nullptr, k16, 0, 0);
    gemm_wmma_kernel<<<gQKV, 128, 0, stream>>>(h16, Wv + l * wqkvStride,
        DD, DD, 6, DD * HSZ, HSZ, nullptr, nullptr, nullptr, vT16, 0, /*vTrans*/1);

    attn_kernel<<<dim3(TT / 64, HH, BB), 128, 0, stream>>>(q16, k16, vT16, x);

    ln_kernel<<<BT, 256, 0, stream>>>(x, ln2_g + l * DD, ln2_b + l * DD, h16);
    gemm_wmma_kernel<<<dim3(DFF / GTN, BT / GTM), 128, 0, stream>>>(
        h16, W1 + (size_t)l * DD * DFF, DD, DFF, 30, 0, DFF,
        b1 + l * DFF, nullptr, nullptr, g16, /*gelu*/1, 0);
    gemm_wmma_kernel<<<dim3(DD / GTN, BT / GTM), 128, 0, stream>>>(
        g16, W2 + (size_t)l * DFF * DD, DFF, DD, 30, 0, DD,
        b2 + l * DD, /*residual*/x, x, nullptr, 0, 0);
  }

  ln_kernel<<<BT, 256, 0, stream>>>(x, lnf_g, lnf_b, h16);
  gemm_wmma_kernel<<<dim3(VV / GTN, BT / GTM), 128, 0, stream>>>(
      h16, Wlm, DD, VV, 30, 0, VV, blm, nullptr, out, nullptr, 0, 0);
}
